// WaveNetModelContinuos_80693845557937
// MI455X (gfx1250) — matrix operand order repaired and checked by reading, not compiled, not run on hardware
//
#include <hip/hip_runtime.h>
#include <hip/hip_bf16.h>

typedef __attribute__((ext_vector_type(16))) __bf16 v16bf;
typedef __attribute__((ext_vector_type(8)))  float  v8f;

#define NLAYER 50
#define NRES   32
#define NSKIP  512
#define NB     4
#define NT     16384
#define TSKIP  5115
#define LOUT   11269
#define LTILES 42          // per-layer packed A tiles: 4 sig + 4 tan + 2 res + 32 skip
#define P1TILES 512        // post1: 32 M-tiles x 16 K-tiles
#define TILE_H 512         // halfs per packed tile (32 lanes x 16)

__device__ __forceinline__ int imin(int a, int b){ return a < b ? a : b; }
__device__ __forceinline__ int imax(int a, int b){ return a > b ? a : b; }

__device__ __forceinline__ __bf16 f2bf(float f){
  unsigned u = __builtin_bit_cast(unsigned, f);
  unsigned r = (u + 0x7FFFu + ((u >> 16) & 1u)) >> 16;   // round-to-nearest-even
  unsigned short s = (unsigned short)r;
  return __builtin_bit_cast(__bf16, s);
}
__device__ __forceinline__ unsigned pk2(float a, float b){
  unsigned short lo = __builtin_bit_cast(unsigned short, f2bf(a));
  unsigned short hi = __builtin_bit_cast(unsigned short, f2bf(b));
  return (unsigned)lo | ((unsigned)hi << 16);
}
__device__ __forceinline__ uint4 pk8(v8f c){
  uint4 v;
  v.x = pk2(c[0], c[1]); v.y = pk2(c[2], c[3]);
  v.z = pk2(c[4], c[5]); v.w = pk2(c[6], c[7]);
  return v;
}
__device__ __forceinline__ float fast_rcp(float x){ return __builtin_amdgcn_rcpf(x); }
// branch-free: sigmoid(x) = 1/(1+e^-x)   (v_exp_f32 + v_rcp_f32, no exec divergence)
__device__ __forceinline__ float fast_sigmoid(float x){ return fast_rcp(1.0f + __expf(-x)); }
// branch-free: tanh(x) = 1 - 2/(e^{2x}+1); saturates correctly to +/-1 for large |x|
__device__ __forceinline__ float fast_tanh(float x){ return 1.0f - 2.0f * fast_rcp(__expf(2.0f * x) + 1.0f); }
__device__ __forceinline__ float eluf(float x){ return x > 0.0f ? x : (__expf(x) - 1.0f); }

__device__ __forceinline__ v8f wmma_bf(v16bf a, v16bf b, v8f c){
  // D = A(16x32 bf16) x B(32x16 bf16) + C(16x16 f32)
  return __builtin_amdgcn_wmma_f32_16x16x32_bf16(false, a, false, b, (short)0, c, false, false);
}
__device__ __forceinline__ v16bf loadA(const __bf16* p){ return *(const v16bf*)p; }
__device__ __forceinline__ v8f load8f(const float* p){
  float4 a = *(const float4*)p, b = *(const float4*)(p + 4);
  v8f c; c[0]=a.x; c[1]=a.y; c[2]=a.z; c[3]=a.w; c[4]=b.x; c[5]=b.y; c[6]=b.z; c[7]=b.w;
  return c;
}
__device__ __forceinline__ void store8f(float* p, v8f c){
  float4 a, b; a.x=c[0]; a.y=c[1]; a.z=c[2]; a.w=c[3]; b.x=c[4]; b.y=c[5]; b.z=c[6]; b.w=c[7];
  *(float4*)p = a; *(float4*)(p + 4) = b;
}
__device__ __forceinline__ v8f biasC(const float* bp, int base){
  v8f c;
  #pragma unroll
  for (int j = 0; j < 8; ++j) c[j] = bp[base + j];
  return c;
}

// ---- pack all A-operand weights into WMMA lane layout (bf16) ----
// A 16x32 k order used here (contiguous, B matches): lane<16: m=lane, k = h
//                      lane>=16: m=lane-16, k = h+16
__global__ void pack_kernel(const float* __restrict__ sig_w, const float* __restrict__ tan_w,
                            const float* __restrict__ res_w, const float* __restrict__ skip_w,
                            const float* __restrict__ post1_w, __bf16* __restrict__ packed){
  int idx = blockIdx.x * blockDim.x + threadIdx.x;
  const int TOTAL = (NLAYER * LTILES + P1TILES) * TILE_H;
  if (idx >= TOTAL) return;
  int tile = idx >> 9;
  int r    = idx & 511;
  int lane = r >> 4;
  int h    = r & 15;
  int m  = lane & 15;
  int hi = lane >> 4;
  int k  = hi * 16 + h;
  float val;
  if (tile < NLAYER * LTILES){
    int layer = tile / LTILES;
    int t = tile % LTILES;
    if (t < 8){                                   // sig (0-3) / tan (4-7): tt = kk*2+mt
      const float* w = (t < 4) ? sig_w : tan_w;
      int tt = t & 3;
      int kk = tt >> 1;                           // conv tap (0: h[u-d], 1: h[u])
      int mt = tt & 1;
      int o = mt * 16 + m;
      val = w[(((layer * NRES + o) * NRES) + k) * 2 + kk];
    } else if (t < 10){                           // res
      int mt = t - 8;
      int o = mt * 16 + m;
      val = res_w[((layer * NRES + o) * NRES) + k];
    } else {                                      // skip
      int mt = t - 10;
      int o = mt * 16 + m;
      val = skip_w[((layer * NSKIP + o) * NRES) + k];
    }
  } else {                                        // post1: tile = mt*16 + kc
    int pt = tile - NLAYER * LTILES;
    int mt = pt >> 4;
    int kc = pt & 15;
    int o = mt * 16 + m;
    val = post1_w[o * NSKIP + kc * 32 + k];
  }
  packed[idx] = f2bf(val);
}

// ---- skip bias totals + skip_sum init ----
__global__ void skipb_kernel(const float* __restrict__ skip_b, float* __restrict__ sbt){
  int o = blockIdx.x * blockDim.x + threadIdx.x;
  if (o >= NSKIP) return;
  float s = 0.0f;
  for (int i = 0; i < NLAYER; ++i) s += skip_b[i * NSKIP + o];
  sbt[o] = s;
}
__global__ void skip_init_kernel(const float* __restrict__ sbt, float* __restrict__ skips){
  size_t idx = (size_t)blockIdx.x * blockDim.x + threadIdx.x;
  const size_t total = (size_t)NB * LOUT * (NSKIP / 4);
  if (idx >= total) return;
  int o4 = (int)(idx & 127);                       // 512 ch = 128 float4 per row
  ((float4*)skips)[idx] = ((const float4*)sbt)[o4];
}

// ---- front 1x1 conv: [B,T,8] -> h0 [B,T,32] (f32 master + bf16 shadow) ----
__global__ void front_kernel(const float* __restrict__ in, const float* __restrict__ from_w,
                             const float* __restrict__ from_b, float* __restrict__ h0,
                             __bf16* __restrict__ h0bf){
  int idx = blockIdx.x * blockDim.x + threadIdx.x;
  if (idx >= NB * NT) return;
  const float* ip = in + (size_t)idx * 8;
  float4 a = *(const float4*)ip, b4 = *(const float4*)(ip + 4);
  float x[8] = {a.x, a.y, a.z, a.w, b4.x, b4.y, b4.z, b4.w};
  float acc[NRES];
  #pragma unroll 4
  for (int o = 0; o < NRES; ++o){
    float s = from_b[o];
    #pragma unroll
    for (int c = 0; c < 8; ++c) s += from_w[o * 8 + c] * x[c];
    acc[o] = s;
  }
  float* op = h0 + (size_t)idx * NRES;
  #pragma unroll
  for (int o = 0; o < NRES; o += 4){
    float4 v; v.x = acc[o]; v.y = acc[o+1]; v.z = acc[o+2]; v.w = acc[o+3];
    *(float4*)(op + o) = v;
  }
  __bf16* bp = h0bf + (size_t)idx * NRES;
  #pragma unroll
  for (int o = 0; o < NRES; o += 8){
    uint4 v;
    v.x = pk2(acc[o],   acc[o+1]); v.y = pk2(acc[o+2], acc[o+3]);
    v.z = pk2(acc[o+4], acc[o+5]); v.w = pk2(acc[o+6], acc[o+7]);
    *(uint4*)(bp + o) = v;
  }
}

// ---- one WaveNet layer: gate -> (res update, skip accumulate) ----
// each wave handles 16 time positions; 8 waves / block; grid (time tiles, B)
__global__ void layer_kernel(const float* __restrict__ hin, float* __restrict__ hout,
                             const __bf16* __restrict__ hbf_in, __bf16* __restrict__ hbf_out,
                             float* __restrict__ skips, const __bf16* __restrict__ pa,
                             const float* __restrict__ sig_b, const float* __restrict__ tan_b,
                             const float* __restrict__ res_b, int d, int ustart){
  __shared__ __align__(32) __bf16 lds[8][16][32];
  int lane = threadIdx.x & 31;
  int wv   = threadIdx.x >> 5;
  int n  = lane & 15;
  int hf = lane >> 4;
  int b  = blockIdx.y;
  int u0 = ustart + (blockIdx.x * 8 + wv) * 16;
  int u  = u0 + n;
  int uc = imin(u, NT - 1);

  const float*  hb  = hin    + (size_t)b * NT * NRES;
  float*        ho  = hout   + (size_t)b * NT * NRES;
  const __bf16* hbb = hbf_in + (size_t)b * NT * NRES;

  // B operands from bf16 shadow: 16 consecutive channels (k = hf*16 + h) at one t
  v16bf Bc = *(const v16bf*)(hbb + (size_t)uc * NRES + hf * 16);        // h[u]   (tap 1)
  v16bf Bm = *(const v16bf*)(hbb + (size_t)(uc - d) * NRES + hf * 16);  // h[u-d] (tap 0)

  const __bf16* al = pa + lane * 16;
  v8f s0 = biasC(sig_b, hf * 8);
  v8f s1 = biasC(sig_b, 16 + hf * 8);
  s0 = wmma_bf(loadA(al + 0 * TILE_H), Bm, s0);
  s1 = wmma_bf(loadA(al + 1 * TILE_H), Bm, s1);
  s0 = wmma_bf(loadA(al + 2 * TILE_H), Bc, s0);
  s1 = wmma_bf(loadA(al + 3 * TILE_H), Bc, s1);
  v8f t0 = biasC(tan_b, hf * 8);
  v8f t1 = biasC(tan_b, 16 + hf * 8);
  t0 = wmma_bf(loadA(al + 4 * TILE_H), Bm, t0);
  t1 = wmma_bf(loadA(al + 5 * TILE_H), Bm, t1);
  t0 = wmma_bf(loadA(al + 6 * TILE_H), Bc, t0);
  t1 = wmma_bf(loadA(al + 7 * TILE_H), Bc, t1);

  // gate = sigmoid(sig)*tanh(tan); stage to LDS (C-layout -> B-layout transpose)
  __bf16* lp = &lds[wv][n][0];
  #pragma unroll
  for (int j = 0; j < 8; ++j)
    lp[hf * 8 + j]      = f2bf(fast_sigmoid(s0[j]) * fast_tanh(t0[j]));
  #pragma unroll
  for (int j = 0; j < 8; ++j)
    lp[16 + hf * 8 + j] = f2bf(fast_sigmoid(s1[j]) * fast_tanh(t1[j]));
  __syncthreads();
  v16bf Bg = *(const v16bf*)&lds[wv][n][hf * 16];

  // residual path: hout[u] = res_w @ gate + res_b + hin[u]  (f32 master + bf16 shadow)
  v8f r0 = biasC(res_b, hf * 8);
  v8f r1 = biasC(res_b, 16 + hf * 8);
  r0 = wmma_bf(loadA(al + 8 * TILE_H), Bg, r0);
  r1 = wmma_bf(loadA(al + 9 * TILE_H), Bg, r1);
  if (u < NT){
    v8f h0v = load8f(hb + (size_t)u * NRES + hf * 8);
    v8f h1v = load8f(hb + (size_t)u * NRES + 16 + hf * 8);
    #pragma unroll
    for (int j = 0; j < 8; ++j){ r0[j] += h0v[j]; r1[j] += h1v[j]; }
    store8f(ho + (size_t)u * NRES + hf * 8, r0);
    store8f(ho + (size_t)u * NRES + 16 + hf * 8, r1);
    __bf16* hbo = hbf_out + ((size_t)b * NT + u) * NRES;
    *(uint4*)(hbo + hf * 8)      = pk8(r0);
    *(uint4*)(hbo + 16 + hf * 8) = pk8(r1);
  }

  // skip path: skip_sum[u] += skip_w @ gate (only for u >= TSKIP)
  if (u0 + 15 >= TSKIP){
    int usk = imin(imax(u, TSKIP), NT - 1);
    float* sp = skips + ((size_t)b * LOUT + (usk - TSKIP)) * NSKIP;
    bool ok = (u >= TSKIP) && (u < NT);
    for (int mt = 0; mt < 32; ++mt){
      float* cp = sp + mt * 16 + hf * 8;
      v8f c = load8f(cp);
      c = wmma_bf(loadA(al + (10 + mt) * TILE_H), Bg, c);
      if (ok) store8f(cp, c);
    }
  }
}

// ---- fused post: elu -> post1(512x512) -> elu -> post2 dot ----
// each wave handles 16 time positions; 2 waves / block (32 KB LDS)
__global__ void post_kernel(const float* __restrict__ skips, const __bf16* __restrict__ pa,
                            const float* __restrict__ post1_b, const float* __restrict__ post2_w,
                            const float* __restrict__ post2_b, float* __restrict__ out){
  __shared__ __align__(32) __bf16 lds[2][16][NSKIP];
  int lane = threadIdx.x & 31;
  int wv   = threadIdx.x >> 5;
  int n  = lane & 15;
  int hf = lane >> 4;
  int b  = blockIdx.y;
  const int ntiles = (LOUT + 15) / 16;
  int tile = blockIdx.x * 2 + wv;
  int tc   = imin(tile, ntiles - 1);
  int t    = tc * 16 + n;
  int tcl  = imin(t, LOUT - 1);
  const float* sp = skips + ((size_t)b * LOUT + tcl) * NSKIP;

  // stage elu(skip_sum) as bf16 [t][512] in LDS (each lane: its row, 256 channels)
  for (int cc = hf * 256; cc < hf * 256 + 256; cc += 8){
    float4 x = *(const float4*)(sp + cc);
    float4 y = *(const float4*)(sp + cc + 4);
    uint4 v;
    v.x = pk2(eluf(x.x), eluf(x.y));
    v.y = pk2(eluf(x.z), eluf(x.w));
    v.z = pk2(eluf(y.x), eluf(y.y));
    v.w = pk2(eluf(y.z), eluf(y.w));
    *(uint4*)&lds[wv][n][cc] = v;
  }
  __syncthreads();

  float psum = 0.0f;
  for (int mt = 0; mt < 32; ++mt){
    v8f acc = biasC(post1_b, mt * 16 + hf * 8);
    #pragma unroll 4
    for (int kc = 0; kc < 16; ++kc){
      v16bf Bk = *(const v16bf*)&lds[wv][n][kc * 32 + hf * 16];
      v16bf Ak = loadA(pa + (size_t)(mt * 16 + kc) * TILE_H + lane * 16);
      acc = wmma_bf(Ak, Bk, acc);
    }
    #pragma unroll
    for (int j = 0; j < 8; ++j)
      psum += post2_w[mt * 16 + hf * 8 + j] * eluf(acc[j]);
  }
  psum += __shfl_xor(psum, 16, 32);   // combine channel halves across half-waves
  if (hf == 0 && tile < ntiles && t < LOUT)
    out[(size_t)b * LOUT + t] = psum + post2_b[0];
}

extern "C" void kernel_launch(void* const* d_in, const int* in_sizes, int n_in,
                              void* d_out, int out_size, void* d_ws, size_t ws_size,
                              hipStream_t stream){
  const float* in      = (const float*)d_in[0];
  const float* from_w  = (const float*)d_in[1];
  const float* from_b  = (const float*)d_in[2];
  const float* sig_w   = (const float*)d_in[3];
  const float* sig_b   = (const float*)d_in[4];
  const float* tan_w   = (const float*)d_in[5];
  const float* tan_b   = (const float*)d_in[6];
  const float* skip_w  = (const float*)d_in[7];
  const float* skip_b  = (const float*)d_in[8];
  const float* res_w   = (const float*)d_in[9];
  const float* res_b   = (const float*)d_in[10];
  const float* post1_w = (const float*)d_in[11];
  const float* post1_b = (const float*)d_in[12];
  const float* post2_w = (const float*)d_in[13];
  const float* post2_b = (const float*)d_in[14];
  float* out = (float*)d_out;

  char* ws = (char*)d_ws;
  const size_t SZ_H    = (size_t)NB * NT * NRES * 4;            // 8 MiB
  const size_t SZ_HBF  = (size_t)NB * NT * NRES * 2;            // 4 MiB
  const size_t SZ_SKIP = (size_t)NB * LOUT * NSKIP * 4;         // ~92 MiB (L2-resident)
  const size_t SZ_PACK = (size_t)(NLAYER * LTILES + P1TILES) * TILE_H * 2;
  float*  hA     = (float*)(ws);
  float*  hB     = (float*)(ws + SZ_H);
  __bf16* hbfA   = (__bf16*)(ws + 2 * SZ_H);
  __bf16* hbfB   = (__bf16*)(ws + 2 * SZ_H + SZ_HBF);
  float*  skips  = (float*)(ws + 2 * SZ_H + 2 * SZ_HBF);
  __bf16* packed = (__bf16*)(ws + 2 * SZ_H + 2 * SZ_HBF + SZ_SKIP);
  float*  sbt    = (float*)(ws + 2 * SZ_H + 2 * SZ_HBF + SZ_SKIP + SZ_PACK);

  {
    int total = (NLAYER * LTILES + P1TILES) * TILE_H;
    pack_kernel<<<(total + 255) / 256, 256, 0, stream>>>(sig_w, tan_w, res_w, skip_w, post1_w, packed);
  }
  skipb_kernel<<<2, 256, 0, stream>>>(skip_b, sbt);
  {
    size_t total = (size_t)NB * LOUT * (NSKIP / 4);
    skip_init_kernel<<<(unsigned)((total + 255) / 256), 256, 0, stream>>>(sbt, skips);
  }
  front_kernel<<<(NB * NT + 255) / 256, 256, 0, stream>>>(in, from_w, from_b, hA, hbfA);

  const float*  hi  = hA;
  float*        ho  = hB;
  const __bf16* hbi = hbfA;
  __bf16*       hbo = hbfB;
  int S = 0;
  for (int i = 0; i < NLAYER; ++i){
    int d = 1 << (i % 10);
    int ustart = S + d;
    int count = NT - ustart;
    int bx = (count + 127) / 128;
    layer_kernel<<<dim3(bx, NB), 256, 0, stream>>>(hi, ho, hbi, hbo, skips,
        packed + (size_t)i * LTILES * TILE_H,
        sig_b + i * NRES, tan_b + i * NRES, res_b + i * NRES, d, ustart);
    { const float*  t1 = ho;  ho  = (float*)hi;   hi  = t1; }
    { const __bf16* t2 = hbo; hbo = (__bf16*)hbi; hbi = t2; }
    S = ustart;
  }

  {
    int ntiles = (LOUT + 15) / 16;
    post_kernel<<<dim3((ntiles + 1) / 2, NB), 64, 0, stream>>>(skips,
        packed + (size_t)NLAYER * LTILES * TILE_H, post1_b, post2_w, post2_b, out);
  }
  (void)in_sizes; (void)n_in; (void)out_size; (void)ws_size;
}
